// StereoCNN_37142877176359
// MI455X (gfx1250) — compile-verified, matrix operand order checked by reading, not run on hardware
//
#include <hip/hip_runtime.h>
#include <hip/hip_bf16.h>
#include <math.h>

// ---------------------------------------------------------------------------
// StereoCNN for MI455X (gfx1250): bf16 WMMA implicit-GEMM convs + banded-GEMM
// correlation. Features stored NHWC bf16 so WMMA A/B fragments are contiguous
// 128-bit loads matching the CDNA5 16-bit matrix VGPR layout.  Halo-tile
// staging uses GLOBAL_LOAD_ASYNC_TO_LDS_B128 (ASYNCcnt) when the toolchain
// exposes the builtin; otherwise falls back to load+ds_store.
// ---------------------------------------------------------------------------

#define HH 384
#define WW 1280
#define CC 64
#define KTOT 576           // 64 ch * 9 taps
#define KDISP 128

#if __has_builtin(__builtin_amdgcn_global_load_async_to_lds_b128)
#define HAVE_ASYNC_LDS 1
#else
#define HAVE_ASYNC_LDS 0
#endif

typedef __attribute__((ext_vector_type(16))) __bf16 v16bf;
typedef __attribute__((ext_vector_type(8)))  __bf16 bf16x8;
typedef __attribute__((ext_vector_type(8)))  float  v8f;
typedef int v4i_vs __attribute__((vector_size(16)));   // builtin's pointee type

__device__ __forceinline__ v16bf mk16(bf16x8 lo, bf16x8 hi) {
  union { struct { bf16x8 a, b; } p; v16bf v; } u;
  u.p.a = lo; u.p.b = hi;
  return u.v;
}

__device__ __forceinline__ v8f wmma_bf16(v16bf a, v16bf b, v8f c) {
  // (neg_a, A, neg_b, B, c_mod, C, reuse_a, reuse_b)
  return __builtin_amdgcn_wmma_f32_16x16x32_bf16(false, a, false, b,
                                                 (short)0, c, false, false);
}

#if HAVE_ASYNC_LDS
__device__ __forceinline__ void async_copy_b128(const __bf16* gsrc, __bf16* ldst) {
  __builtin_amdgcn_global_load_async_to_lds_b128(
      (__attribute__((address_space(1))) v4i_vs*)gsrc,
      (__attribute__((address_space(3))) v4i_vs*)ldst, 0, 0);
}
__device__ __forceinline__ void wait_async_done() {
#if __has_builtin(__builtin_amdgcn_s_wait_asynccnt)
  __builtin_amdgcn_s_wait_asynccnt(0);
#else
  asm volatile("s_wait_asynccnt 0x0" ::: "memory");
#endif
}
#endif

// ---------------------------------------------------------------------------
// Layer 0: 3 -> 64, tanh.  Small K (27): plain FMA, weights cached in LDS.
// Output NHWC bf16: feat[ib][y][x][c], ib = img*2 + batch (0,1=left 2,3=right)
// ---------------------------------------------------------------------------
__global__ __launch_bounds__(256) void conv0_kernel(
    const float* __restrict__ l, const float* __restrict__ r,
    const float* __restrict__ w0, const float* __restrict__ b0,
    __bf16* __restrict__ fout)
{
  __shared__ float wb[64 * 27];
  __shared__ float bb[64];
  const int tid = threadIdx.x;
  for (int i = tid; i < 64 * 27; i += 256) wb[i] = w0[i];   // [o][ci][ky][kx]
  if (tid < 64) bb[tid] = b0[tid];
  __syncthreads();

  const int x  = blockIdx.x * 256 + tid;   // 5*256 == 1280 exactly
  const int y  = blockIdx.y;
  const int ib = blockIdx.z;
  const int img = ib >> 1, bidx = ib & 1;
  const float* src = (img ? r : l) + (size_t)bidx * 3 * HH * WW;

  float patch[27];
#pragma unroll
  for (int c = 0; c < 3; ++c)
#pragma unroll
    for (int dy = 0; dy < 3; ++dy)
#pragma unroll
      for (int dx = 0; dx < 3; ++dx) {
        const int gy = y + dy - 1, gx = x + dx - 1;
        patch[(c * 3 + dy) * 3 + dx] =
            (gy >= 0 && gy < HH && gx >= 0 && gx < WW)
                ? src[((size_t)c * HH + gy) * WW + gx] : 0.0f;
      }

  __bf16* o = fout + (((size_t)ib * HH + y) * WW + x) * CC;
  for (int oc = 0; oc < 64; ++oc) {
    float a = bb[oc];
#pragma unroll
    for (int k = 0; k < 27; ++k) a = fmaf(wb[oc * 27 + k], patch[k], a);
    o[oc] = (__bf16)tanhf(a);
  }
}

// ---------------------------------------------------------------------------
// Layers 1..3: 64 -> 64 implicit GEMM with v_wmma_f32_16x16x32_bf16.
//   M = 64 pixels along x (one row), N = 32 out-channels per block (z-split),
//   K = 576 (tap-major, channel-minor) -> 18 WMMA K-steps of 32.
//   8 waves = 4(M) x 2(N) 16x16 tiles.  8-row y strip amortizes weight stage.
// LDS: wT[32][576] bf16 (36,864B) + halo tile [3][66][64] bf16 (25,344B).
// ---------------------------------------------------------------------------
__global__ __launch_bounds__(256) void conv_mid_kernel(
    const __bf16* __restrict__ fin, const float* __restrict__ w,
    const float* __restrict__ bias, __bf16* __restrict__ fout, int do_tanh)
{
  __shared__ __bf16 wbuf[32 * KTOT];        // wT[local_o][k]
  __shared__ __bf16 inbuf[3 * 66 * CC];     // [row][xpos][ci]
  __shared__ float  sbias[32];

  const int tid    = threadIdx.x;
  const int x0     = blockIdx.x * 64;
  const int ystrip = blockIdx.y * 8;
  const int ib     = blockIdx.z >> 1;
  const int nblk   = (blockIdx.z & 1) * 32;        // which 32 out-channels

  const __bf16* finb  = fin  + (size_t)ib * HH * WW * CC;
  __bf16*       foutb = fout + (size_t)ib * HH * WW * CC;

  // Stage weights transposed+converted: wbuf[lo*576 + tap*64 + ci]
  for (int idx = tid; idx < 32 * KTOT; idx += 256) {
    const int lo = idx / KTOT, k = idx % KTOT;
    const int tap = k >> 6, ci = k & 63;
    const int ky = tap / 3, kx = tap - ky * 3;
    const int o = nblk + lo;
    wbuf[idx] = (__bf16)w[((o * 64 + ci) * 3 + ky) * 3 + kx];
  }
  if (tid < 32) sbias[tid] = bias[nblk + tid];
  __syncthreads();

  const int lane = tid & 31;
  const int wid  = tid >> 5;
  const int g    = lane >> 4;       // K half-block select within a wave
  const int ml   = lane & 15;       // matrix row / col within a tile
  const int m0   = (wid >> 1) * 16; // wave M tile: 0,16,32,48
  const int nw   = (wid & 1) * 16;  // wave N tile within the 32-ch block

  for (int yy = 0; yy < 8; ++yy) {
    const int y = ystrip + yy;

    // Stage halo tile: rows y-1..y+1, x0-1..x0+64, 64ch, zero-padded (SAME).
    // In-bounds 16B chunks go memory->LDS via the async data mover path
    // (no VGPR round-trip, tracked by ASYNCcnt); OOB chunks are zero-filled.
    for (int c = tid; c < 3 * 66 * 8; c += 256) {
      const int pos = c >> 3, cg = (c & 7) * 8;
      const int rrow = pos / 66, xx = pos % 66;
      const int gy = y - 1 + rrow, gx = x0 - 1 + xx;
      __bf16* dst = inbuf + (size_t)(rrow * 66 + xx) * CC + cg;
      if (gy >= 0 && gy < HH && gx >= 0 && gx < WW) {
        const __bf16* srcp = finb + ((size_t)gy * WW + gx) * CC + cg;
#if HAVE_ASYNC_LDS
        async_copy_b128(srcp, dst);
#else
        *(bf16x8*)dst = *(const bf16x8*)srcp;
#endif
      } else {
        *(bf16x8*)dst = (bf16x8){};
      }
    }
#if HAVE_ASYNC_LDS
    wait_async_done();
#endif
    __syncthreads();

    if (yy < 7)   // pull next row's cacheline path early (global_prefetch_b8)
      __builtin_prefetch(finb + ((size_t)(y + 2) * WW + x0) * CC, 0, 1);

    v8f acc = {};
#pragma unroll
    for (int kb = 0; kb < 18; ++kb) {
      const int tap = kb >> 1;
      const int ch  = (kb & 1) * 32;
      const int ky  = tap / 3, kx = tap - ky * 3;

      // A fragment: per-lane two b128 LDS loads (K = g*8+0..7 and +16..)
      const __bf16* ap =
          inbuf + ((size_t)(ky * 66 + (m0 + ml + kx))) * CC + ch + g * 8;
      const v16bf a = mk16(*(const bf16x8*)ap, *(const bf16x8*)(ap + 16));

      // B fragment: weight row for out-channel (nw+ml), same K pattern
      const __bf16* bp = wbuf + (size_t)(nw + ml) * KTOT + kb * 32 + g * 8;
      const v16bf b = mk16(*(const bf16x8*)bp, *(const bf16x8*)(bp + 16));

      acc = wmma_bf16(a, b, acc);
    }

    // C layout: element r -> (m = r + g*8, n = lane&15)
#pragma unroll
    for (int rr = 0; rr < 8; ++rr) {
      const int m = rr + g * 8;
      const int n = ml;
      float v = acc[rr] + sbias[nw + n];
      if (do_tanh) v = tanhf(v);
      foutb[((size_t)y * WW + (x0 + m0 + m)) * CC + nblk + nw + n] = (__bf16)v;
    }
    __syncthreads();
  }
}

// ---------------------------------------------------------------------------
// Correlation: banded GEMM.  Per (b,y):  C[x, x'] = sum_c L[x][c] * R[x'][c],
// d = x - x' in [0,128).  Block = one 16-x tile; wave t handles x' tile
// x'0 = X0 - 16t (t = 0..8); K = 64 -> two bf16 WMMAs.  x' < 0 rows are zero.
// ---------------------------------------------------------------------------
__global__ __launch_bounds__(288) void corr_kernel(
    const __bf16* __restrict__ fl, const __bf16* __restrict__ fr,
    float* __restrict__ out)
{
  const int lane = threadIdx.x & 31;
  const int t    = threadIdx.x >> 5;        // 0..8 diagonal-band tile
  const int g    = lane >> 4;
  const int ml   = lane & 15;
  const int X0   = blockIdx.x * 16;
  const int y    = blockIdx.y;
  const int b    = blockIdx.z;

  const __bf16* L = fl + ((size_t)b * HH + y) * (size_t)WW * CC;
  const __bf16* R = fr + ((size_t)b * HH + y) * (size_t)WW * CC;
  const int Xp0 = X0 - t * 16;

  v8f acc = {};
#pragma unroll
  for (int ks = 0; ks < 2; ++ks) {
    const int cb = ks * 32;

    const __bf16* ap = L + (size_t)(X0 + ml) * CC + cb + g * 8;
    const v16bf a = mk16(*(const bf16x8*)ap, *(const bf16x8*)(ap + 16));

    const int xp = Xp0 + ml;                // B row (n = lane&15)
    bf16x8 blo = {}, bhi = {};
    if (xp >= 0) {
      const __bf16* bp = R + (size_t)xp * CC + cb + g * 8;
      blo = *(const bf16x8*)bp;
      bhi = *(const bf16x8*)(bp + 16);
    }
    const v16bf bm = mk16(blo, bhi);        // waves reconverge before WMMA

    acc = wmma_bf16(a, bm, acc);
  }

#pragma unroll
  for (int rr = 0; rr < 8; ++rr) {
    const int m = rr + g * 8;
    const int n = ml;
    const int d = m - n + t * 16;           // disparity for this element
    if (d >= 0 && d < KDISP) {
      const int x = X0 + m;
      out[(((size_t)b * KDISP + d) * HH + y) * WW + x] = acc[rr];
    }
  }
}

// ---------------------------------------------------------------------------
// Launch: conv0 -> buf0; L1 buf0->buf1 (tanh); L2 buf1->buf0 (tanh);
//         L3 buf0->buf1 (linear); corr(buf1 left, buf1 right) -> d_out.
// Workspace: two NHWC bf16 buffers of 4*384*1280*64 elems (2 x 252 MB).
// ---------------------------------------------------------------------------
extern "C" void kernel_launch(void* const* d_in, const int* in_sizes, int n_in,
                              void* d_out, int out_size, void* d_ws, size_t ws_size,
                              hipStream_t stream) {
  (void)in_sizes; (void)n_in; (void)out_size; (void)ws_size;
  const float* l  = (const float*)d_in[0];
  const float* r  = (const float*)d_in[1];
  const float* W0 = (const float*)d_in[2];
  const float* B0 = (const float*)d_in[3];
  const float* W1 = (const float*)d_in[4];
  const float* B1 = (const float*)d_in[5];
  const float* W2 = (const float*)d_in[6];
  const float* B2 = (const float*)d_in[7];
  const float* W3 = (const float*)d_in[8];
  const float* B3 = (const float*)d_in[9];
  float* out = (float*)d_out;

  const size_t plane = (size_t)HH * WW * CC;   // per-ib feature plane
  __bf16* buf0 = (__bf16*)d_ws;
  __bf16* buf1 = buf0 + 4 * plane;

  conv0_kernel<<<dim3(WW / 256, HH, 4), 256, 0, stream>>>(l, r, W0, B0, buf0);

  const dim3 cg(WW / 64, HH / 8, 8);           // 8 = 4 ib * 2 out-channel halves
  conv_mid_kernel<<<cg, 256, 0, stream>>>(buf0, W1, B1, buf1, 1);
  conv_mid_kernel<<<cg, 256, 0, stream>>>(buf1, W2, B2, buf0, 1);
  conv_mid_kernel<<<cg, 256, 0, stream>>>(buf0, W3, B3, buf1, 0);

  corr_kernel<<<dim3(WW / 16, HH, 2), 288, 0, stream>>>(
      buf1 /* left ib 0,1 */, buf1 + 2 * plane /* right ib 2,3 */, out);
}
